// DeepSeekV32MLA_71133248356444
// MI455X (gfx1250) — matrix operand order repaired and checked by reading, not compiled, not run on hardware
//
#include <hip/hip_runtime.h>

// ---------------------------------------------------------------------------
// DeepSeek V3.2 MLA + lightning indexer for MI455X (gfx1250, wave32, WMMA)
// All GEMM-shaped math runs through v_wmma_f32_16x16x32_bf16 (f32 accum).
// Staging is vectorized: global_load_b128 -> cvt -> ds_store_b128.
// Optional CDNA5 paths (compile-time guarded): TDM tensor_load_to_lds for the
// top-k score row, global_load_async_to_lds_b32 for the top-k index stage.
// ---------------------------------------------------------------------------

typedef __attribute__((ext_vector_type(16))) __bf16 v16bf;
typedef __attribute__((ext_vector_type(8)))  __bf16 bf16x8;
typedef __attribute__((ext_vector_type(8)))  float  v8f;

#define T_    1024
#define DIM_  2048
#define H_    16
#define QLR_  1024
#define KVLR_ 512
#define NOPE_ 128
#define ROPE_ 64
#define VD_   128
#define QKD_  192
#define IH_   16
#define ID_   128
#define TOPK_ 256

// ---- WMMA fragment loaders (contiguous k = 16*(lane/16) + s, same for A and B) -----------------
__device__ __forceinline__ v16bf load_a_frag(const __bf16* rowptr, int lane) {
  int hb = lane >> 4;
  bf16x8 a0 = *(const bf16x8*)(rowptr + hb * 16);
  bf16x8 a1 = *(const bf16x8*)(rowptr + hb * 16 + 8);
  v16bf r;
#pragma unroll
  for (int i = 0; i < 8; ++i) { r[i] = a0[i]; r[i + 8] = a1[i]; }
  return r;
}
__device__ __forceinline__ v16bf load_b_frag(const __bf16* rowptr, int lane) {
  int hb = lane >> 4;
  bf16x8 b0 = *(const bf16x8*)(rowptr + hb * 16);
  bf16x8 b1 = *(const bf16x8*)(rowptr + hb * 16 + 8);
  v16bf r;
#pragma unroll
  for (int i = 0; i < 8; ++i) { r[i] = b0[i]; r[i + 8] = b1[i]; }
  return r;
}
__device__ __forceinline__ v8f wmma_bf16(v16bf a, v16bf b, v8f c) {
  return __builtin_amdgcn_wmma_f32_16x16x32_bf16(false, a, false, b, (short)0, c,
                                                 false, false);
}

// ---- vectorized staging: 16 contiguous f32 -> 16 bf16 in LDS --------------
__device__ __forceinline__ void stage16(const float* __restrict__ g, __bf16* l) {
  const float4* s = (const float4*)g;
  float4 f0 = s[0], f1 = s[1], f2 = s[2], f3 = s[3];
  bf16x8 lo, hi;
  lo[0] = (__bf16)f0.x; lo[1] = (__bf16)f0.y; lo[2] = (__bf16)f0.z; lo[3] = (__bf16)f0.w;
  lo[4] = (__bf16)f1.x; lo[5] = (__bf16)f1.y; lo[6] = (__bf16)f1.z; lo[7] = (__bf16)f1.w;
  hi[0] = (__bf16)f2.x; hi[1] = (__bf16)f2.y; hi[2] = (__bf16)f2.z; hi[3] = (__bf16)f2.w;
  hi[4] = (__bf16)f3.x; hi[5] = (__bf16)f3.y; hi[6] = (__bf16)f3.z; hi[7] = (__bf16)f3.w;
  *(bf16x8*)l = lo;
  *(bf16x8*)(l + 8) = hi;
}
__device__ __forceinline__ void stage16_zero(__bf16* l) {
  bf16x8 z;
#pragma unroll
  for (int i = 0; i < 8; ++i) z[i] = (__bf16)0.0f;
  *(bf16x8*)l = z;
  *(bf16x8*)(l + 8) = z;
}

// ---- optional CDNA5 extras ------------------------------------------------
#if defined(__AMDGCN__) && __has_builtin(__builtin_amdgcn_tensor_load_to_lds)
#define HAVE_TDM 1
typedef unsigned int u32x4 __attribute__((ext_vector_type(4)));
typedef int i32x4 __attribute__((ext_vector_type(4)));
typedef int i32x8 __attribute__((ext_vector_type(8)));
// 1-D tile copy: nelem f32 from global -> LDS (D# per CDNA5 ISA)
__device__ __forceinline__ void tdm_load_row_f32(void* lds_dst,
                                                 const float* gsrc, int nelem) {
  unsigned lds_off =
      (unsigned)(unsigned long long)(__attribute__((address_space(3))) char*)lds_dst;
  unsigned long long ga = (unsigned long long)gsrc;
  u32x4 g0;
  g0[0] = 1u;                                   // count=1 (valid user D#)
  g0[1] = lds_off;                              // lds_addr
  g0[2] = (unsigned)(ga & 0xffffffffu);         // global_addr[31:0]
  g0[3] = (unsigned)((ga >> 32) & 0x1ffffffu)   // global_addr[56:32]
          | (2u << 30);                         // type=2 ("image")
  i32x8 g1;
  g1[0] = (int)(2u << 16);                      // data_size=4B; wg_mask=0
  g1[1] = (int)(((unsigned)nelem & 0xffffu) << 16);  // tensor_dim0[15:0]
  g1[2] = (int)((unsigned)nelem >> 16);              // tensor_dim0[31:16]
  g1[3] = (int)(((unsigned)nelem & 0xffffu) << 16);  // tile_dim0
  g1[4] = 0;                                    // tile_dim1/2 unused
  g1[5] = nelem;                                // tensor_dim0_stride[31:0]
  g1[6] = 0;
  g1[7] = 0;
  i32x4 z4;
  z4[0] = z4[1] = z4[2] = z4[3] = 0;
#if __clang_major__ >= 23
  i32x8 z8;
#pragma unroll
  for (int i = 0; i < 8; ++i) z8[i] = 0;
  __builtin_amdgcn_tensor_load_to_lds(g0, g1, z4, z4, z8, 0);
#else
  __builtin_amdgcn_tensor_load_to_lds(g0, g1, z4, z4, 0);
#endif
}
#else
#define HAVE_TDM 0
#endif

#if defined(__AMDGCN__) && \
    __has_builtin(__builtin_amdgcn_global_load_async_to_lds_b32)
#define HAVE_ASYNC_LDS 1
typedef __attribute__((address_space(1))) int* gint_ptr;
typedef __attribute__((address_space(3))) int* lint_ptr;
#else
#define HAVE_ASYNC_LDS 0
#endif

// ---------------------------------------------------------------------------
// Generic GEMM: C[M,N] = A[M,K] @ Bl^T, Bl logical NxK.
//   b_trans=1 : Bl[n][k] = B[n*ldb + k]   (C = A @ B.T, weight [N,K])
//   b_trans=0 : Bl[n][k] = B[k*ldb + n]   (C = A @ B,   weight [K,N])
// Tile 128x128x32, 8 waves, each wave owns a 32x64 sub-tile (8 WMMA / K-step).
// M multiple of 128, K multiple of 32; N guarded.
// ---------------------------------------------------------------------------
__global__ __launch_bounds__(256) void gemm_kernel(
    const float* __restrict__ A, const float* __restrict__ B,
    float* __restrict__ C, int M, int N, int K, int lda, int ldb, int ldc,
    int b_trans) {
  __shared__ __bf16 As[128][32];
  __shared__ __bf16 Bs[128][32];
  const int mb = blockIdx.y * 128;
  const int nb = blockIdx.x * 128;
  const int tid = threadIdx.x, lane = tid & 31, wv = tid >> 5;
  const int wm = (wv & 3) * 32;   // wave row offset in tile
  const int wn = (wv >> 2) * 64;  // wave col offset in tile

  v8f acc[2][4];
#pragma unroll
  for (int mt = 0; mt < 2; ++mt)
#pragma unroll
    for (int nt = 0; nt < 4; ++nt)
#pragma unroll
      for (int j = 0; j < 8; ++j) acc[mt][nt][j] = 0.0f;

  const int r2 = tid >> 1;            // 0..127
  const int cg = (tid & 1) << 4;      // 0 or 16

  for (int kb = 0; kb < K; kb += 32) {
    // stage A tile (128x32), one 16-float chunk per thread
    stage16(&A[(size_t)(mb + r2) * lda + kb + cg], &As[r2][cg]);
    // stage B tile (128 n x 32 k)
    if (b_trans) {
      int n = nb + r2;
      if (n < N)
        stage16(&B[(size_t)n * ldb + kb + cg], &Bs[r2][cg]);
      else
        stage16_zero(&Bs[r2][cg]);
    } else {
      int k = tid >> 3;               // 0..31
      int ncg = (tid & 7) << 4;       // 0..112
#pragma unroll
      for (int j = 0; j < 16; ++j) {
        int n = nb + ncg + j;
        float v = (n < N) ? B[(size_t)(kb + k) * ldb + n] : 0.0f;
        Bs[ncg + j][k] = (__bf16)v;
      }
    }
    if (kb + 32 < K)  // global_prefetch_b8 for next A tile
      __builtin_prefetch(&A[(size_t)(mb + r2) * lda + kb + 32 + cg], 0, 1);
    __syncthreads();

    v16bf af0 = load_a_frag(&As[wm + (lane & 15)][0], lane);
    v16bf af1 = load_a_frag(&As[wm + 16 + (lane & 15)][0], lane);
#pragma unroll
    for (int nt = 0; nt < 4; ++nt) {
      v16bf bfr = load_b_frag(&Bs[wn + nt * 16 + (lane & 15)][0], lane);
      acc[0][nt] = wmma_bf16(af0, bfr, acc[0][nt]);
      acc[1][nt] = wmma_bf16(af1, bfr, acc[1][nt]);
    }
    __syncthreads();
  }
#pragma unroll
  for (int mt = 0; mt < 2; ++mt)
#pragma unroll
    for (int nt = 0; nt < 4; ++nt)
#pragma unroll
      for (int j = 0; j < 8; ++j) {
        int m = mb + wm + mt * 16 + j + 8 * (lane >> 4);
        int n = nb + wn + nt * 16 + (lane & 15);
        if (n < N) C[(size_t)m * ldc + n] = acc[mt][nt][j];
      }
}

// ---------------------------------------------------------------------------
// RMSNorm rows of length C (eps 1e-6)
// ---------------------------------------------------------------------------
__global__ __launch_bounds__(256) void rmsnorm_kernel(const float* __restrict__ x,
                                                      const float* __restrict__ w,
                                                      float* __restrict__ y, int C) {
  __shared__ float red[256];
  int t = blockIdx.x, tid = threadIdx.x;
  const float* p = x + (size_t)t * C;
  float ss = 0.0f;
  for (int i = tid; i < C; i += 256) { float v = p[i]; ss += v * v; }
  red[tid] = ss;
  __syncthreads();
  for (int s = 128; s > 0; s >>= 1) {
    if (tid < s) red[tid] += red[tid + s];
    __syncthreads();
  }
  float inv = rsqrtf(red[0] * (1.0f / C) + 1e-6f);
  float* o = y + (size_t)t * C;
  for (int i = tid; i < C; i += 256) o[i] = p[i] * inv * w[i];
}

// ---------------------------------------------------------------------------
// FWHT-64 on wave32 (5 xor-shuffle stages + 1 in-lane stage)
// ---------------------------------------------------------------------------
__device__ __forceinline__ void fwht64(float& lo, float& hi, int L) {
#pragma unroll
  for (int h = 1; h < 32; h <<= 1) {
    float plo = __shfl_xor(lo, h, 32);
    float phi = __shfl_xor(hi, h, 32);
    lo = (L & h) ? (plo - lo) : (lo + plo);
    hi = (L & h) ? (phi - hi) : (hi + phi);
  }
  float a = lo + hi, b = lo - hi;
  lo = a;
  hi = b;
}

// ---------------------------------------------------------------------------
// Indexer post-proc: optional layernorm(128) -> rope(first 64) ->
// rotate_activation (out = [FWHT64(x2), FWHT64(x1)]).  One wave per (t,h).
// ---------------------------------------------------------------------------
__global__ __launch_bounds__(32) void indexer_post_kernel(
    const float* __restrict__ src, float* __restrict__ dst,
    const float* __restrict__ cosT, const float* __restrict__ sinT,
    const float* __restrict__ lnw, const float* __restrict__ lnb, int do_ln) {
  int h = blockIdx.x, t = blockIdx.y, L = threadIdx.x;
  const float* p = src + ((size_t)t * 16 + h) * 128;
  float e0 = p[L], e1 = p[L + 32], e2 = p[L + 64], e3 = p[L + 96];
  if (do_ln) {
    float s = e0 + e1 + e2 + e3;
    for (int sh = 16; sh > 0; sh >>= 1) s += __shfl_xor(s, sh, 32);
    float mean = s * (1.0f / 128.0f);
    float d0 = e0 - mean, d1 = e1 - mean, d2 = e2 - mean, d3 = e3 - mean;
    float v = d0 * d0 + d1 * d1 + d2 * d2 + d3 * d3;
    for (int sh = 16; sh > 0; sh >>= 1) v += __shfl_xor(v, sh, 32);
    float inv = rsqrtf(v * (1.0f / 128.0f) + 1e-5f);
    e0 = d0 * inv * lnw[L] + lnb[L];
    e1 = d1 * inv * lnw[L + 32] + lnb[L + 32];
    e2 = d2 * inv * lnw[L + 64] + lnb[L + 64];
    e3 = d3 * inv * lnw[L + 96] + lnb[L + 96];
  }
  float c = cosT[t * 32 + L], s = sinT[t * 32 + L];
  float r0 = e0 * c - e1 * s;
  float r1 = e0 * s + e1 * c;
  e0 = r0; e1 = r1;
  fwht64(e2, e3, L);  // x2 half -> out[0..63]
  fwht64(e0, e1, L);  // x1 half -> out[64..127]
  float* o = dst + ((size_t)t * 16 + h) * 128;
  o[L] = e2; o[L + 32] = e3; o[L + 64] = e0; o[L + 96] = e1;
}

// ---------------------------------------------------------------------------
// Indexer scores: score[t,s] = mask( sum_h relu(iq[t,h]·ik[s,h]) * w[t,h]*sc )
// ---------------------------------------------------------------------------
__global__ __launch_bounds__(256) void score_kernel(
    const float* __restrict__ iq, const float* __restrict__ ik,
    const float* __restrict__ wts, float* __restrict__ score) {
  __shared__ __bf16 Aq[128][ID_];  // 32 KB
  __shared__ __bf16 Bk[64][ID_];   // 16 KB
  const int tb = blockIdx.y * 128;
  const int sb = blockIdx.x * 64;
  const int tid = threadIdx.x, lane = tid & 31, wv = tid >> 5;

  v8f acc[4];
#pragma unroll
  for (int nt = 0; nt < 4; ++nt)
#pragma unroll
    for (int j = 0; j < 8; ++j) acc[nt][j] = 0.0f;

  for (int h = 0; h < IH_; ++h) {
    for (int i = tid; i < 128 * 8; i += 256) {
      int r = i >> 3, cgi = (i & 7) << 4;
      stage16(&iq[((size_t)(tb + r) * 16 + h) * 128 + cgi], &Aq[r][cgi]);
    }
    for (int i = tid; i < 64 * 8; i += 256) {
      int r = i >> 3, cgi = (i & 7) << 4;
      stage16(&ik[((size_t)(sb + r) * 16 + h) * 128 + cgi], &Bk[r][cgi]);
    }
    __syncthreads();

    v8f hacc[4];
#pragma unroll
    for (int nt = 0; nt < 4; ++nt)
#pragma unroll
      for (int j = 0; j < 8; ++j) hacc[nt][j] = 0.0f;

    for (int kb = 0; kb < ID_; kb += 32) {
      v16bf af = load_a_frag(&Aq[wv * 16 + (lane & 15)][kb], lane);
#pragma unroll
      for (int nt = 0; nt < 4; ++nt) {
        v16bf bfr = load_b_frag(&Bk[nt * 16 + (lane & 15)][kb], lane);
        hacc[nt] = wmma_bf16(af, bfr, hacc[nt]);
      }
    }
#pragma unroll
    for (int nt = 0; nt < 4; ++nt)
#pragma unroll
      for (int j = 0; j < 8; ++j) {
        int m = wv * 16 + j + 8 * (lane >> 4);
        float ww = wts[(size_t)(tb + m) * 16 + h];
        float v = hacc[nt][j];
        acc[nt][j] += (v > 0.0f ? v : 0.0f) * ww;
      }
    __syncthreads();
  }
  const float sc = 0.022097086912079608f;  // IH^-0.5 * ID^-0.5
#pragma unroll
  for (int nt = 0; nt < 4; ++nt)
#pragma unroll
    for (int j = 0; j < 8; ++j) {
      int m = tb + wv * 16 + j + 8 * (lane >> 4);
      int n = sb + nt * 16 + (lane & 15);
      float v = acc[nt][j] * sc;
      if (n > m) v = -3e30f;  // causal mask
      score[(size_t)m * T_ + n] = v;
    }
}

// ---------------------------------------------------------------------------
// Top-256 per row via iterative block argmax. Invalid picks stored as -1.
// Score row staged to LDS via TDM when available.
// ---------------------------------------------------------------------------
__global__ __launch_bounds__(256) void topk_kernel(const float* __restrict__ score,
                                                   int* __restrict__ topk) {
  __shared__ float vals[T_];
  __shared__ float rv[256];
  __shared__ int ri[256];
  int t = blockIdx.x, tid = threadIdx.x;
#if HAVE_TDM
  if (tid < 32) tdm_load_row_f32(vals, score + (size_t)t * T_, T_);
  asm volatile("s_wait_tensorcnt 0x0" ::: "memory");
  __syncthreads();
#else
  for (int i = tid; i < T_; i += 256) vals[i] = score[(size_t)t * T_ + i];
  __syncthreads();
#endif
  for (int k = 0; k < TOPK_; ++k) {
    float bv = -1e38f;
    int bi = 0;
    for (int i = tid; i < T_; i += 256) {
      float v = vals[i];
      if (v > bv) { bv = v; bi = i; }
    }
    rv[tid] = bv; ri[tid] = bi;
    __syncthreads();
    for (int s = 128; s > 0; s >>= 1) {
      if (tid < s && rv[tid + s] > rv[tid]) { rv[tid] = rv[tid + s]; ri[tid] = ri[tid + s]; }
      __syncthreads();
    }
    if (tid == 0) {
      int b = ri[0];
      topk[t * TOPK_ + k] = (rv[0] > -1e29f) ? b : -1;
      vals[b] = -2e30f;
    }
    __syncthreads();
  }
}

// ---------------------------------------------------------------------------
// kv_a (T,576) -> kv_key (T,576): rmsnorm first 512, rope last 64
// ---------------------------------------------------------------------------
__global__ __launch_bounds__(256) void kv_post_kernel(
    const float* __restrict__ kva, const float* __restrict__ w,
    const float* __restrict__ cosT, const float* __restrict__ sinT,
    float* __restrict__ kvkey) {
  __shared__ float red[256];
  int t = blockIdx.x, tid = threadIdx.x;
  const float* x = kva + (size_t)t * 576;
  float a = x[tid], b = x[tid + 256];
  red[tid] = a * a + b * b;
  __syncthreads();
  for (int s = 128; s > 0; s >>= 1) {
    if (tid < s) red[tid] += red[tid + s];
    __syncthreads();
  }
  float inv = rsqrtf(red[0] * (1.0f / 512.0f) + 1e-6f);
  kvkey[(size_t)t * 576 + tid] = a * inv * w[tid];
  kvkey[(size_t)t * 576 + tid + 256] = b * inv * w[tid + 256];
  if (tid < 32) {
    float c = cosT[t * 32 + tid], s = sinT[t * 32 + tid];
    float x1 = x[512 + tid], x2 = x[544 + tid];
    kvkey[(size_t)t * 576 + 512 + tid] = x1 * c - x2 * s;
    kvkey[(size_t)t * 576 + 544 + tid] = x1 * s + x2 * c;
  }
}

// ---------------------------------------------------------------------------
// q (T,16,192): rope on cols 128..191 -> write to qkey cols 512..575
// ---------------------------------------------------------------------------
__global__ __launch_bounds__(32) void q_rope_kernel(const float* __restrict__ q,
                                                    float* __restrict__ qkey,
                                                    const float* __restrict__ cosT,
                                                    const float* __restrict__ sinT) {
  int h = blockIdx.x, t = blockIdx.y, L = threadIdx.x;
  const float* p = q + ((size_t)t * 16 + h) * QKD_ + NOPE_;
  float c = cosT[t * 32 + L], s = sinT[t * 32 + L];
  float x1 = p[L], x2 = p[L + 32];
  float* o = qkey + ((size_t)t * 16 + h) * 576 + 512;
  o[L] = x1 * c - x2 * s;
  o[L + 32] = x1 * s + x2 * c;
}

// ---------------------------------------------------------------------------
// Sparse attention per token t (256 thr = 8 waves):
//   logits(16x256) = qkey_t @ gather(kv_key)^T * 192^-0.5   (WMMA)
//   softmax rows, out_latent(16x512) = attn @ gather(kv_key[:, :512]) (WMMA)
// ---------------------------------------------------------------------------
__global__ __launch_bounds__(256) void attn_kernel(
    const float* __restrict__ qkey, const float* __restrict__ kvkey,
    const int* __restrict__ topk, float* __restrict__ olat) {
  __shared__ __bf16 Qs[16][576];
  __shared__ float Lg[16][256];
  __shared__ __bf16 At[16][256];
  __shared__ int idx[256];
  int t = blockIdx.x, tid = threadIdx.x, lane = tid & 31, wv = tid >> 5;

  for (int i = tid; i < 16 * 36; i += 256) {
    int h = i / 36, cgi = (i % 36) << 4;
    stage16(&qkey[((size_t)t * 16 + h) * 576 + cgi], &Qs[h][cgi]);
  }
#if HAVE_ASYNC_LDS
  __builtin_amdgcn_global_load_async_to_lds_b32(
      (gint_ptr)(topk + t * TOPK_ + tid), (lint_ptr)&idx[tid], 0, 0);
  asm volatile("s_wait_asynccnt 0x0" ::: "memory");
#else
  idx[tid] = topk[t * TOPK_ + tid];
#endif
  __syncthreads();

  // ---- logits: wave wv handles key cols [wv*32, wv*32+32)
  v8f lacc[2];
#pragma unroll
  for (int nt = 0; nt < 2; ++nt)
#pragma unroll
    for (int j = 0; j < 8; ++j) lacc[nt][j] = 0.0f;

  for (int kb = 0; kb < 576; kb += 32) {
    v16bf af = load_a_frag(&Qs[lane & 15][kb], lane);
#pragma unroll
    for (int nt = 0; nt < 2; ++nt) {
      int n = wv * 32 + nt * 16 + (lane & 15);
      int j = idx[n];
      int sj = j < 0 ? 0 : j;
      const float* bp = kvkey + (size_t)sj * 576 + kb + ((lane >> 4) << 4);
      v16bf bfr;
#pragma unroll
      for (int q4 = 0; q4 < 4; ++q4) {
        float4 f = ((const float4*)bp)[q4];
        bfr[q4 * 4 + 0] = (__bf16)f.x;
        bfr[q4 * 4 + 1] = (__bf16)f.y;
        bfr[q4 * 4 + 2] = (__bf16)f.z;
        bfr[q4 * 4 + 3] = (__bf16)f.w;
      }
      lacc[nt] = wmma_bf16(af, bfr, lacc[nt]);
    }
  }
  const float scale = 0.07216878364870323f;  // 192^-0.5
#pragma unroll
  for (int nt = 0; nt < 2; ++nt)
#pragma unroll
    for (int j = 0; j < 8; ++j) {
      int h = j + 8 * (lane >> 4);
      int n = wv * 32 + nt * 16 + (lane & 15);
      float v = lacc[nt][j] * scale;
      if (idx[n] < 0) v = -1e30f;
      Lg[h][n] = v;
    }
  __syncthreads();

  // ---- softmax: wave wv handles heads 2wv, 2wv+1
  for (int r = 0; r < 2; ++r) {
    int h = wv * 2 + r;
    float m = -1e30f;
    for (int i = lane; i < 256; i += 32) m = fmaxf(m, Lg[h][i]);
    for (int s = 16; s > 0; s >>= 1) m = fmaxf(m, __shfl_xor(m, s, 32));
    float sum = 0.0f;
    for (int i = lane; i < 256; i += 32) {
      float e = __expf(Lg[h][i] - m);
      sum += e;
      Lg[h][i] = e;
    }
    for (int s = 16; s > 0; s >>= 1) sum += __shfl_xor(sum, s, 32);
    float inv = 1.0f / sum;
    for (int i = lane; i < 256; i += 32) At[h][i] = (__bf16)(Lg[h][i] * inv);
  }
  __syncthreads();

  // ---- out_latent: wave wv handles latent cols [wv*64, wv*64+64)
  v8f oacc[4];
#pragma unroll
  for (int nt = 0; nt < 4; ++nt)
#pragma unroll
    for (int j = 0; j < 8; ++j) oacc[nt][j] = 0.0f;

  for (int kb = 0; kb < 256; kb += 32) {
    v16bf af = load_a_frag(&At[lane & 15][kb], lane);
    int hb16 = (lane >> 4) << 4;
#pragma unroll
    for (int nt = 0; nt < 4; ++nt) {
      int cb = wv * 64 + nt * 16 + (lane & 15);
      v16bf bfr;
#pragma unroll
      for (int i = 0; i < 16; ++i) {
        int j = idx[kb + hb16 + i];
        int sj = j < 0 ? 0 : j;  // attn weight ~0 for invalid anyway
        bfr[i] = (__bf16)kvkey[(size_t)sj * 576 + cb];
      }
      oacc[nt] = wmma_bf16(af, bfr, oacc[nt]);
    }
  }
#pragma unroll
  for (int nt = 0; nt < 4; ++nt)
#pragma unroll
    for (int j = 0; j < 8; ++j) {
      int h = j + 8 * (lane >> 4);
      int c = wv * 64 + nt * 16 + (lane & 15);
      olat[((size_t)t * 16 + h) * 512 + c] = oacc[nt][j];
    }
}

// ---------------------------------------------------------------------------
extern "C" void kernel_launch(void* const* d_in, const int* in_sizes, int n_in,
                              void* d_out, int out_size, void* d_ws,
                              size_t ws_size, hipStream_t stream) {
  (void)in_sizes; (void)n_in; (void)out_size; (void)ws_size;
  const float* x        = (const float*)d_in[0];
  const float* rope_cos = (const float*)d_in[1];
  const float* rope_sin = (const float*)d_in[2];
  const float* wq_a     = (const float*)d_in[3];
  const float* q_norm_w = (const float*)d_in[4];
  const float* wq_b     = (const float*)d_in[5];
  const float* wkv_a    = (const float*)d_in[6];
  const float* kv_norm_w= (const float*)d_in[7];
  const float* wkv_b    = (const float*)d_in[8];
  const float* wo       = (const float*)d_in[9];
  const float* iq_norm_w= (const float*)d_in[10];
  const float* iwq_b    = (const float*)d_in[11];
  const float* iwk      = (const float*)d_in[12];
  const float* k_norm_w = (const float*)d_in[13];
  const float* k_norm_b = (const float*)d_in[14];
  const float* wproj    = (const float*)d_in[15];
  float* out = (float*)d_out;

  float* ws = (float*)d_ws;
  size_t off = 0;
  auto alloc = [&](size_t n) { float* p = ws + off; off += n; return p; };
  float* qr    = alloc((size_t)T_ * QLR_);        // 1M f32
  float* qniq  = alloc((size_t)T_ * QLR_);        // 1M
  float* qnq   = alloc((size_t)T_ * QLR_);        // 1M
  float* iq    = alloc((size_t)T_ * IH_ * ID_);   // 2M
  float* ik    = alloc((size_t)T_ * IH_ * ID_);   // 2M
  float* wts   = alloc((size_t)T_ * IH_);         // 16K
  float* score = alloc((size_t)T_ * T_);          // 1M
  float* qbuf  = alloc((size_t)T_ * H_ * QKD_);   // 3M
  float* kva   = alloc((size_t)T_ * 576);
  float* kvkey = alloc((size_t)T_ * 576);
  int*   topk  = (int*)alloc((size_t)T_ * TOPK_);
  float* qkey  = alloc((size_t)T_ * H_ * 576);    // 9M
  // dead-buffer reuse: out_latent (8M) over [qr..score] (8.015M),
  // out_v (2M) over qbuf (3M)
  float* olat = qr;
  float* outv = qbuf;

  dim3 blk(256);
  // qr = x @ wq_a.T
  gemm_kernel<<<dim3(QLR_ / 128, T_ / 128), blk, 0, stream>>>(
      x, wq_a, qr, T_, QLR_, DIM_, DIM_, DIM_, QLR_, 1);
  rmsnorm_kernel<<<T_, 256, 0, stream>>>(qr, iq_norm_w, qniq, QLR_);
  rmsnorm_kernel<<<T_, 256, 0, stream>>>(qr, q_norm_w, qnq, QLR_);
  // iq_pre = rms(qr) @ iwq_b.T
  gemm_kernel<<<dim3(2048 / 128, T_ / 128), blk, 0, stream>>>(
      qniq, iwq_b, iq, T_, 2048, QLR_, QLR_, QLR_, 2048, 1);
  // ik_pre = x @ iwk.T
  gemm_kernel<<<dim3(2048 / 128, T_ / 128), blk, 0, stream>>>(
      x, iwk, ik, T_, 2048, DIM_, DIM_, DIM_, 2048, 1);
  // wts = x @ wproj.T   (N=16, guarded)
  gemm_kernel<<<dim3(1, T_ / 128), blk, 0, stream>>>(
      x, wproj, wts, T_, IH_, DIM_, DIM_, DIM_, IH_, 1);
  // indexer post: iq (no LN), ik (LN)
  indexer_post_kernel<<<dim3(IH_, T_), 32, 0, stream>>>(
      iq, iq, rope_cos, rope_sin, nullptr, nullptr, 0);
  indexer_post_kernel<<<dim3(IH_, T_), 32, 0, stream>>>(
      ik, ik, rope_cos, rope_sin, k_norm_w, k_norm_b, 1);
  // scores + causal mask
  score_kernel<<<dim3(T_ / 64, T_ / 128), blk, 0, stream>>>(iq, ik, wts, score);
  // top-256
  topk_kernel<<<T_, 256, 0, stream>>>(score, topk);
  // q = rms(qr) @ wq_b.T
  gemm_kernel<<<dim3(3072 / 128, T_ / 128), blk, 0, stream>>>(
      qnq, wq_b, qbuf, T_, H_ * QKD_, QLR_, QLR_, QLR_, H_ * QKD_, 1);
  // kv_a = x @ wkv_a.T
  gemm_kernel<<<dim3((576 + 127) / 128, T_ / 128), blk, 0, stream>>>(
      x, wkv_a, kva, T_, 576, DIM_, DIM_, DIM_, 576, 1);
  kv_post_kernel<<<T_, 256, 0, stream>>>(kva, kv_norm_w, rope_cos, rope_sin, kvkey);
  // q_nope_proj per head: qkey[:, h, :512] = q_nope_h @ w_k_h  (NN)
  for (int h = 0; h < H_; ++h) {
    gemm_kernel<<<dim3(KVLR_ / 128, T_ / 128), blk, 0, stream>>>(
        qbuf + (size_t)h * QKD_, wkv_b + (size_t)h * 256 * KVLR_,
        qkey + (size_t)h * 576, T_, KVLR_, NOPE_, H_ * QKD_, KVLR_, H_ * 576, 0);
  }
  q_rope_kernel<<<dim3(H_, T_), 32, 0, stream>>>(qbuf, qkey, rope_cos, rope_sin);
  // sparse attention -> out_latent
  attn_kernel<<<T_, 256, 0, stream>>>(qkey, kvkey, topk, olat);
  // out_v per head: outv[:, h, :] = olat_h @ w_v_h.T
  for (int h = 0; h < H_; ++h) {
    gemm_kernel<<<dim3(1, T_ / 128), blk, 0, stream>>>(
        olat + (size_t)h * KVLR_, wkv_b + ((size_t)h * 256 + NOPE_) * KVLR_,
        outv + (size_t)h * VD_, T_, VD_, KVLR_, H_ * KVLR_, KVLR_, H_ * VD_, 1);
  }
  // final: out = outv @ wo.T
  gemm_kernel<<<dim3(DIM_ / 128, T_ / 128), blk, 0, stream>>>(
      outv, wo, out, T_, DIM_, H_ * VD_, H_ * VD_, H_ * VD_, DIM_, 1);
}
